// Attention_11836929868027
// MI455X (gfx1250) — matrix operand order repaired and checked by reading, not compiled, not run on hardware
//
#include <hip/hip_runtime.h>

// Problem constants (from reference)
#define Bsz 2
#define Tq 1024
#define Dm 4096
#define NH 32
#define KHN 8
#define HD 128
#define QKV_LD 6144   // NH*HD + 2*KHN*HD
#define KOFF 4096     // column offset of k inside fused qkv row
#define VOFF 5120     // column offset of v

typedef __bf16 bf16;
typedef __attribute__((ext_vector_type(16))) __bf16 bf16x16;
typedef __attribute__((ext_vector_type(8)))  float  f32x8;
typedef unsigned int u32;

// ---------------- bf16 helpers (round-to-nearest-even) ----------------
__device__ __forceinline__ unsigned short bf_bits(float f) {
  union { float f; u32 u; } c; c.f = f;
  return (unsigned short)((c.u + 0x7FFFu + ((c.u >> 16) & 1u)) >> 16);
}
__device__ __forceinline__ bf16 f2bf(float f) {
  union { unsigned short s; bf16 b; } o; o.s = bf_bits(f);
  return o.b;
}
__device__ __forceinline__ float bf2f(bf16 b) {
  union { bf16 b; unsigned short s; } i; i.b = b;
  union { u32 u; float f; } o; o.u = ((u32)i.s) << 16;
  return o.f;
}
__device__ __forceinline__ void st_bf16x4(bf16* dst, float4 v) {
  union { unsigned long long u; unsigned short s[4]; } pk;
  pk.s[0] = bf_bits(v.x); pk.s[1] = bf_bits(v.y);
  pk.s[2] = bf_bits(v.z); pk.s[3] = bf_bits(v.w);
  *reinterpret_cast<unsigned long long*>(dst) = pk.u;   // -> ds_store_b64
}
__device__ __forceinline__ f32x8 zero8() {
  f32x8 z;
#pragma unroll
  for (int i = 0; i < 8; ++i) z[i] = 0.f;
  return z;
}

// ---------------- CDNA5 async global->LDS copy (ASYNCcnt path) ----------------
// 16B per lane, byte-for-byte; LDS offset = low 32 bits of generic shared ptr.
__device__ __forceinline__ void async_ld16(void* lds_dst, const void* gsrc) {
  u32 lds_off = (u32)(uintptr_t)lds_dst;
  asm volatile("global_load_async_to_lds_b128 %0, %1, off"
               :: "v"(lds_off), "v"(gsrc)
               : "memory");
}
__device__ __forceinline__ void wait_async0() {
  asm volatile("s_wait_asynccnt 0x0" ::: "memory");
}

// ---------------- WMMA fragment helpers (ISA 7.12.2 layouts) ----------------
__device__ __forceinline__ bf16x16 load_a_frag(const bf16* base, int ld) {
  const int lane = threadIdx.x & 31;
  const bf16* p = base + (lane & 15) * ld + (lane >> 4) * 8;
  bf16x16 a;
#pragma unroll
  for (int e = 0; e < 8; ++e)  a[e] = p[e];
#pragma unroll
  for (int e = 8; e < 16; ++e) a[e] = p[e + 8];
  return a;
}
// B frag from N-major LDS tile (Bt[n][k]): col = lane&15, K = 8*(lane>>4)+{0..7, 16..23} per lane
__device__ __forceinline__ bf16x16 load_b_frag(const bf16* base, int ld) {
  const int lane = threadIdx.x & 31;
  const bf16* p = base + (lane & 15) * ld + (lane >> 4) * 8;
  bf16x16 b;
#pragma unroll
  for (int e = 0; e < 16; ++e) b[e] = p[e < 8 ? e : e + 8];
  return b;
}
__device__ __forceinline__ f32x8 wmma_bf16(bf16x16 a, bf16x16 b, f32x8 c) {
  return __builtin_amdgcn_wmma_f32_16x16x32_bf16(false, a, false, b, (short)0, c, false, false);
}
// C/D frag: M = v + 8*(lane>>4), N = lane&15
__device__ __forceinline__ void store_c_frag(float* dst, int ld, f32x8 c) {
  const int lane = threadIdx.x & 31;
  const int col = lane & 15, hlf = lane >> 4;
#pragma unroll
  for (int v = 0; v < 8; ++v) dst[(v + 8 * hlf) * ld + col] = c[v];
}
__device__ __forceinline__ void store_c_frag(bf16* dst, int ld, f32x8 c) {
  const int lane = threadIdx.x & 31;
  const int col = lane & 15, hlf = lane >> 4;
#pragma unroll
  for (int v = 0; v < 8; ++v) dst[(v + 8 * hlf) * ld + col] = f2bf(c[v]);
}

// ---------------- 128x128 GEMM tile (K-step 32), bf16 WMMA ----------------
// AT: float (VALU-convert staging) or bf16 (pure async global->LDS copy).
template <typename AT, typename CT>
__device__ __forceinline__ void gemm_tile_128x128(
    const AT* __restrict__ A, int lda,
    const float* __restrict__ Bw, int ldb,
    CT* __restrict__ C, int ldc, int K)
{
  __shared__ bf16 As[128 * 32];   // [m][k], row-major, 8KB
  __shared__ bf16 Bs[128 * 32];   // [n][k], N-major (transposed), 8KB
  const int tid = threadIdx.x;
  const int wid = tid >> 5;
  const int wm  = wid & 3;        // 4 waves along M -> 32 rows each
  const int wn  = wid >> 2;       // 2 waves along N -> 64 cols each

  f32x8 acc[2][4];
#pragma unroll
  for (int i = 0; i < 2; ++i)
#pragma unroll
    for (int j = 0; j < 4; ++j) acc[i][j] = zero8();

  for (int kb = 0; kb < K; kb += 32) {
    __syncthreads();
    // A tile: 128x32
    if constexpr (sizeof(AT) == 2) {
      // bf16 source: async byte-copy, 16B (8 elems) per lane
#pragma unroll
      for (int p = 0; p < 2; ++p) {
        int idx = p * 256 + tid;
        int row = idx >> 2, c8 = (idx & 3) * 8;
        async_ld16(&As[row * 32 + c8], &A[row * lda + kb + c8]);
      }
    } else {
      // f32 source: float4 along K, convert, packed bf16x4 LDS stores
#pragma unroll
      for (int p = 0; p < 4; ++p) {
        int idx = p * 256 + tid;
        int row = idx >> 3, c4 = (idx & 7) * 4;
        float4 v = *reinterpret_cast<const float4*>(&A[row * lda + kb + c4]);
        st_bf16x4(&As[row * 32 + c4], v);
      }
    }
    // B tile (f32 weights): float4 along N, convert + transpose into Bs[n][k]
#pragma unroll
    for (int p = 0; p < 4; ++p) {
      int idx = p * 256 + tid;
      int k = idx >> 5, n4 = (idx & 31) * 4;
      float4 v = *reinterpret_cast<const float4*>(&Bw[(kb + k) * ldb + n4]);
      Bs[(n4 + 0) * 32 + k] = f2bf(v.x);
      Bs[(n4 + 1) * 32 + k] = f2bf(v.y);
      Bs[(n4 + 2) * 32 + k] = f2bf(v.z);
      Bs[(n4 + 3) * 32 + k] = f2bf(v.w);
    }
    if (kb + 32 < K) __builtin_prefetch(&Bw[(kb + 32) * ldb + tid], 0, 1);
    if constexpr (sizeof(AT) == 2) wait_async0();
    __syncthreads();

    bf16x16 af[2], bfr[4];
#pragma unroll
    for (int i = 0; i < 2; ++i) af[i]  = load_a_frag(&As[(wm * 32 + i * 16) * 32], 32);
#pragma unroll
    for (int j = 0; j < 4; ++j) bfr[j] = load_b_frag(&Bs[(wn * 64 + j * 16) * 32], 32);
#pragma unroll
    for (int i = 0; i < 2; ++i)
#pragma unroll
      for (int j = 0; j < 4; ++j)
        acc[i][j] = wmma_bf16(af[i], bfr[j], acc[i][j]);
  }
#pragma unroll
  for (int i = 0; i < 2; ++i)
#pragma unroll
    for (int j = 0; j < 4; ++j)
      store_c_frag(&C[(wm * 32 + i * 16) * ldc + wn * 64 + j * 16], ldc, acc[i][j]);
}

// ---------------- Kernel 0: first non-zero segment index per batch ----------------
__global__ void k_start(const int* __restrict__ seg, int* __restrict__ startb) {
  if (threadIdx.x == 0) {
    const int b = blockIdx.x;
    int s = 0;
    for (int t = 0; t < Tq; ++t) { if (seg[b * Tq + t] != 0) break; ++s; }
    startb[b] = s;
  }
}

// ---------------- Kernel 1: fused QKV projection GEMM (f32 in -> bf16 out) ----------------
__global__ __launch_bounds__(256) void k_gemm_qkv(
    const float* __restrict__ x, const float* __restrict__ wq,
    const float* __restrict__ wk, const float* __restrict__ wv,
    bf16* __restrict__ qkvraw)
{
  const int mt = blockIdx.x;
  const int nb = blockIdx.y * 128;           // fused column base (q|k|v)
  const float* Bw; int ldb;
  if (nb < KOFF)      { Bw = wq + nb;          ldb = NH * HD;  }
  else if (nb < VOFF) { Bw = wk + (nb - KOFF); ldb = KHN * HD; }
  else                { Bw = wv + (nb - VOFF); ldb = KHN * HD; }
  gemm_tile_128x128(x + (size_t)mt * 128 * Dm, Dm, Bw, ldb,
                    qkvraw + (size_t)mt * 128 * QKV_LD + nb, QKV_LD, Dm);
}

// ---------------- Kernel 2: in-place RMSNorm + scale + RoPE (bf16 rows) ----------------
// 1/sqrt(HD) logit scale folded into the q heads here (commutes with rotation).
__global__ __launch_bounds__(64) void k_norm_rope(
    bf16* __restrict__ qkv, const float* __restrict__ q_scale,
    const float* __restrict__ k_scale, const int* __restrict__ seg,
    const int* __restrict__ startb)
{
  const int bt = blockIdx.x;                 // b*Tq + t
  const int hidx = blockIdx.y;               // 0..NH-1: q head; NH..NH+KHN-1: k head
  const int b = bt >> 10, t = bt & (Tq - 1);
  const bool is_q = hidx < NH;
  bf16* row = qkv + (size_t)bt * QKV_LD + (is_q ? hidx * HD : KOFF + (hidx - NH) * HD);
  const float* scale = is_q ? q_scale : k_scale;
  const float extra = is_q ? 0.08838834764831845f : 1.0f;   // 1/sqrt(HD)
  const int i = threadIdx.x;                 // 0..63 -> pairs (i, i+64)
  float x1 = bf2f(row[i]), x2 = bf2f(row[i + 64]);
  float ss = x1 * x1 + x2 * x2;
  ss += __shfl_xor(ss, 1, 32);
  ss += __shfl_xor(ss, 2, 32);
  ss += __shfl_xor(ss, 4, 32);
  ss += __shfl_xor(ss, 8, 32);
  ss += __shfl_xor(ss, 16, 32);
  __shared__ float sred[2];
  if ((i & 31) == 0) sred[i >> 5] = ss;
  __syncthreads();
  ss = sred[0] + sred[1];
  const float rinv = rsqrtf(ss * (1.0f / 128.0f) + 1e-6f) * extra;
  const float y1 = x1 * rinv * scale[i];
  const float y2 = x2 * rinv * scale[i + 64];
  const int sg = seg[bt];
  const int pos = (sg != 0) ? (t - startb[b]) : 0;
  const float fr = (float)(2 * i) * (1.0f / 128.0f);
  const float inv_freq = __powf(1.0e6f, -fr);
  const float ang = (float)pos * inv_freq;
  const float sn = __sinf(ang), cs = __cosf(ang);
  row[i]      = f2bf(y1 * cs - y2 * sn);
  row[i + 64] = f2bf(y2 * cs + y1 * sn);
}

// ---------------- Kernel 3: causal GQA flash attention (WMMA + async LDS feeds) ----------------
__global__ __launch_bounds__(256) void k_flash(
    const bf16* __restrict__ qkv, const int* __restrict__ startb,
    bf16* __restrict__ att)
{
  __shared__ __align__(16) unsigned char smem[65536];
  bf16* Qs  = (bf16*)smem;                 // [128][128] bf16 (32KB), recycled as Ps [128][64]
  bf16* Ks  = (bf16*)(smem + 32768);       // [64][128]  ([kv][H], B-operand layout for S1)
  bf16* Vts = (bf16*)(smem + 49152);       // [128][64]  V transposed: [H][kv]
  bf16* Ps  = Qs;

  const int q0  = blockIdx.x * 128;
  const int n   = blockIdx.y;
  const int b   = blockIdx.z;
  const int kvh = n >> 2;                  // GQA: 4 q heads per kv head
  const int tid = threadIdx.x, lane = tid & 31, wid = tid >> 5;
  const int hlf = lane >> 4, c15 = lane & 15;
  const int start = startb[b];

  // Stage Q tile [128][128]: pure async byte-copy (scale pre-folded in k_norm_rope)
#pragma unroll
  for (int p = 0; p < 8; ++p) {
    int idx = p * 256 + tid;
    int row = idx >> 4, c8 = (idx & 15) * 8;
    async_ld16(&Qs[row * 128 + c8],
               &qkv[(size_t)(b * Tq + q0 + row) * QKV_LD + n * HD + c8]);
  }
  wait_async0();
  __syncthreads();
  bf16x16 qa[4];                           // wave's 16 q rows, full H=128 (K-dim)
#pragma unroll
  for (int kk = 0; kk < 4; ++kk)
    qa[kk] = load_a_frag(&Qs[(wid * 16) * 128 + kk * 32], 128);
  __syncthreads();                         // Qs now reusable as Ps

  f32x8 o[8];
#pragma unroll
  for (int f = 0; f < 8; ++f) o[f] = zero8();
  float m_run[8], l_run[8];
#pragma unroll
  for (int v = 0; v < 8; ++v) { m_run[v] = -1e30f; l_run[v] = 0.f; }

  const int nkb = (q0 >> 6) + 2;           // causal: kv blocks of 64 up to q0+128
  for (int kb = 0; kb < nkb; ++kb) {
    const int s0 = kb * 64;
    __syncthreads();
    // K tile [64][128]: async byte-copy (4 chunks/thread)
#pragma unroll
    for (int p = 0; p < 4; ++p) {
      int idx = p * 256 + tid;
      int r = idx >> 4, c8 = (idx & 15) * 8;
      async_ld16(&Ks[r * 128 + c8],
                 &qkv[(size_t)(b * Tq + s0 + r) * QKV_LD + KOFF + kvh * HD + c8]);
    }
    // V tile: bf16x4 loads, transposed scatter into Vts[h][kv]
    unsigned short* VtsU = (unsigned short*)Vts;
#pragma unroll
    for (int p = 0; p < 8; ++p) {
      int idx = p * 256 + tid;
      int r = idx >> 5, c4 = (idx & 31) * 4;
      union { uint2 v; unsigned short s[4]; } pk;
      pk.v = *reinterpret_cast<const uint2*>(
          &qkv[(size_t)(b * Tq + s0 + r) * QKV_LD + VOFF + kvh * HD + c4]);
      VtsU[(c4 + 0) * 64 + r] = pk.s[0];
      VtsU[(c4 + 1) * 64 + r] = pk.s[1];
      VtsU[(c4 + 2) * 64 + r] = pk.s[2];
      VtsU[(c4 + 3) * 64 + r] = pk.s[3];
    }
    wait_async0();
    __syncthreads();

    // S = Q @ K^T : [16 q][64 kv] per wave
    f32x8 sf[4];
#pragma unroll
    for (int j = 0; j < 4; ++j) {
      sf[j] = zero8();
#pragma unroll
      for (int kk = 0; kk < 4; ++kk)
        sf[j] = wmma_bf16(qa[kk], load_b_frag(&Ks[(j * 16) * 128 + kk * 32], 128), sf[j]);
    }
    // Causal + start mask
#pragma unroll
    for (int j = 0; j < 4; ++j) {
      const int sc = s0 + j * 16 + c15;    // key position
#pragma unroll
      for (int v = 0; v < 8; ++v) {
        const int tq = q0 + wid * 16 + v + 8 * hlf;
        if (sc > tq || sc < start) sf[j][v] = -1e30f;
      }
    }
    // Online softmax
#pragma unroll
    for (int v = 0; v < 8; ++v) {
      float rm = fmaxf(fmaxf(sf[0][v], sf[1][v]), fmaxf(sf[2][v], sf[3][v]));
      rm = fmaxf(rm, __shfl_xor(rm, 1, 32));
      rm = fmaxf(rm, __shfl_xor(rm, 2, 32));
      rm = fmaxf(rm, __shfl_xor(rm, 4, 32));
      rm = fmaxf(rm, __shfl_xor(rm, 8, 32));
      const float mn  = fmaxf(m_run[v], rm);
      const float scl = __expf(m_run[v] - mn);
      m_run[v] = mn;
      float rs = 0.f;
      const int qloc = wid * 16 + v + 8 * hlf;
#pragma unroll
      for (int j = 0; j < 4; ++j) {
        const float pp = __expf(sf[j][v] - mn);
        rs += pp;
        Ps[qloc * 64 + j * 16 + c15] = f2bf(pp);
      }
      rs += __shfl_xor(rs, 1, 32);
      rs += __shfl_xor(rs, 2, 32);
      rs += __shfl_xor(rs, 4, 32);
      rs += __shfl_xor(rs, 8, 32);
      l_run[v] = l_run[v] * scl + rs;
#pragma unroll
      for (int f = 0; f < 8; ++f) o[f][v] *= scl;
    }
    // O += P @ V  (P rows are wave-private; same-wave LDS ops are in-order)
    bf16x16 pa[2];
#pragma unroll
    for (int kk = 0; kk < 2; ++kk)
      pa[kk] = load_a_frag(&Ps[(wid * 16) * 64 + kk * 32], 64);
#pragma unroll
    for (int f = 0; f < 8; ++f)
#pragma unroll
      for (int kk = 0; kk < 2; ++kk)
        o[f] = wmma_bf16(pa[kk], load_b_frag(&Vts[(f * 16) * 64 + kk * 32], 64), o[f]);
  }

  // Normalize and write [B*T, N*H] bf16
#pragma unroll
  for (int v = 0; v < 8; ++v) {
    const float inv = 1.f / l_run[v];
    const int row = b * Tq + q0 + wid * 16 + v + 8 * hlf;
#pragma unroll
    for (int f = 0; f < 8; ++f)
      att[(size_t)row * (NH * HD) + n * HD + f * 16 + c15] = f2bf(o[f][v] * inv);
  }
}

// ---------------- Kernel 4: output projection (bf16 A via async, f32 out) ----------------
__global__ __launch_bounds__(256) void k_gemm_out(
    const bf16* __restrict__ att, const float* __restrict__ wo,
    float* __restrict__ out)
{
  gemm_tile_128x128(att + (size_t)blockIdx.x * 128 * (NH * HD), NH * HD,
                    wo + blockIdx.y * 128, Dm,
                    out + (size_t)blockIdx.x * 128 * Dm + blockIdx.y * 128, Dm,
                    NH * HD);
}

// ---------------- Launch ----------------
extern "C" void kernel_launch(void* const* d_in, const int* in_sizes, int n_in,
                              void* d_out, int out_size, void* d_ws, size_t ws_size,
                              hipStream_t stream) {
  const float* x       = (const float*)d_in[0];
  const float* wq      = (const float*)d_in[1];
  const float* wk      = (const float*)d_in[2];
  const float* wv      = (const float*)d_in[3];
  const float* wo      = (const float*)d_in[4];
  const float* q_scale = (const float*)d_in[5];
  const float* k_scale = (const float*)d_in[6];
  const int*   seg     = (const int*)d_in[9];
  float* out = (float*)d_out;

  bf16* qkvraw = (bf16*)d_ws;                                   // [2048][6144] bf16 (24MB)
  bf16* att    = qkvraw + (size_t)(Bsz * Tq) * QKV_LD;          // [2048][4096] bf16 (16MB)
  int*  startb = (int*)(att + (size_t)(Bsz * Tq) * (NH * HD));  // [B] int

  k_start<<<dim3(Bsz), dim3(32), 0, stream>>>(seg, startb);
  k_gemm_qkv<<<dim3((Bsz * Tq) / 128, QKV_LD / 128), dim3(256), 0, stream>>>(
      x, wq, wk, wv, qkvraw);
  k_norm_rope<<<dim3(Bsz * Tq, NH + KHN), dim3(64), 0, stream>>>(
      qkvraw, q_scale, k_scale, seg, startb);
  k_flash<<<dim3(Tq / 128, NH, Bsz), dim3(256), 0, stream>>>(qkvraw, startb, att);
  k_gemm_out<<<dim3((Bsz * Tq) / 128, Dm / 128), dim3(256), 0, stream>>>(att, wo, out);

  (void)in_sizes; (void)n_in; (void)out_size; (void)ws_size;
}
